// Unet2_68289980006753
// MI455X (gfx1250) — matrix operand order repaired and checked by reading, not compiled, not run on hardware
//
#include <hip/hip_runtime.h>

typedef _Float16 half_t;
typedef __attribute__((ext_vector_type(16))) _Float16 v16h;
typedef __attribute__((ext_vector_type(8)))  float    v8f;

#define NB   8      // batch
#define CH   128    // feature dim
#define NQV  25     // NQ
#define KPAD 256    // padded K for adjacency GEMM

static __device__ __forceinline__ long gtid()    { return (long)blockIdx.x * blockDim.x + threadIdx.x; }
static __device__ __forceinline__ long gstride() { return (long)gridDim.x * blockDim.x; }

// ---------------------------------------------------------------------------
// Generic WMMA GEMM:  C[m,n] = sum_k A[m,k] * W[n,k] (+ bias[n])
// A: (M x K) f16 row-major (lda), W: (Nn x K) f16 row-major (ldw).
// Each wave computes a 16 x (16*NT) strip: the A fragment is loaded once per
// 32-wide K step and reused for NT WMMAs (register blocking along N).
// All NT B fragments are loaded into registers BEFORE the WMMA chain so the
// loads issue as one clause and the NT WMMAs pipeline back-to-back.
// Optional batched operation via blockIdx.y with element strides sA/sW/sC.
// ---------------------------------------------------------------------------
template<int NT>
__global__ void __launch_bounds__(256) wmma_gemm_kernel(
    const half_t* __restrict__ A, const half_t* __restrict__ W,
    const float* __restrict__ bias, float* __restrict__ Cf, half_t* __restrict__ Ch,
    int M, int Nn, int K, int lda, int ldw, int ldc,
    long sA, long sW, long sC)
{
  int b = blockIdx.y;
  A += (long)b * sA;
  W += (long)b * sW;
  long coff = (long)b * sC;

  int tilesN  = (Nn + 15) >> 4;
  int tilesM  = (M + 15) >> 4;
  int nstrips = (tilesN + NT - 1) / NT;
  long wave = (long)blockIdx.x * (blockDim.x >> 5) + (threadIdx.x >> 5);
  if (wave >= (long)tilesM * nstrips) return;   // wave-uniform: EXEC stays all-ones

  int mt  = (int)(wave / nstrips);
  int ns0 = (int)(wave % nstrips) * NT;
  int lane = threadIdx.x & 31;
  int hi   = lane >> 4;        // lane group 0..15 / 16..31
  int row  = lane & 15;

  int m = mt * 16 + row; if (m >= M) m = M - 1;          // clamped (store-masked)

  // A-layout (16x32 f16): lanes0-15 hold K in {kb..kb+7, kb+16..kb+23},
  //                       lanes16-31 hold K in {kb+8..kb+15, kb+24..kb+31}.
  const half_t* ap = A + (long)m * lda + hi * 8;

  // B-layout (32x16 f16): lanes0-15 hold K in {kb..kb+7, kb+16..kb+23},
  //                       lanes16-31 hold K in {kb+8..kb+15, kb+24..kb+31}.
  int n[NT];
  const half_t* wp[NT];
#pragma unroll
  for (int i = 0; i < NT; ++i) {
    n[i] = (ns0 + i) * 16 + row;
    int wn = (n[i] < Nn) ? n[i] : (Nn - 1);              // clamped (store-masked)
    wp[i] = W + (long)wn * ldw + hi * 8;
  }

  v8f acc[NT];
#pragma unroll
  for (int i = 0; i < NT; ++i) acc[i] = (v8f){};

  for (int kb = 0; kb < K; kb += 32) {
    // load A fragment + ALL NT B fragments first (one load clause, one wait),
    // then run the NT WMMAs back-to-back so they pipeline in the matrix unit.
    v16h av;
#pragma unroll
    for (int t = 0; t < 8; ++t) { av[t] = ap[kb + t]; av[8 + t] = ap[kb + 16 + t]; }
    v16h bv[NT];
#pragma unroll
    for (int i = 0; i < NT; ++i) {
#pragma unroll
      for (int t = 0; t < 8; ++t) { bv[i][t] = wp[i][kb + t]; bv[i][8 + t] = wp[i][kb + 16 + t]; }
    }
    if (kb + 32 < K) __builtin_prefetch(ap + kb + 32, 0, 1);
#pragma unroll
    for (int i = 0; i < NT; ++i) {
      acc[i] = __builtin_amdgcn_wmma_f32_16x16x32_f16(
          /*neg_a=*/false, av, /*neg_b=*/false, bv[i],
          /*c_mod=*/(short)0, acc[i], /*reuse_a=*/false, /*reuse_b=*/false);
    }
  }

  // C/D layout: VGPR r -> M = r + 8*hi (per lane group), N = lane&15.
#pragma unroll
  for (int i = 0; i < NT; ++i) {
    int wn = (n[i] < Nn) ? n[i] : (Nn - 1);
    float bn = bias ? bias[wn] : 0.0f;
#pragma unroll
    for (int r = 0; r < 8; ++r) {
      int mm = mt * 16 + r + hi * 8;
      if (mm < M && n[i] < Nn) {
        float v = acc[i][r] + bn;
        long o = coff + (long)mm * ldc + n[i];
        if (Cf) Cf[o] = v;
        if (Ch) Ch[o] = (half_t)v;
      }
    }
  }
}

// ---------------------------------------------------------------------------
// h0[b,i,j,c] = |X[b,i,c] - X[b,j,c]|  (f32 -> f16)
// ---------------------------------------------------------------------------
__global__ void pair_abs_kernel(const float* __restrict__ X, half_t* __restrict__ H,
                                int N, int C)
{
  long total = (long)NB * N * N * C;
  for (long idx = gtid(); idx < total; idx += gstride()) {
    int c = (int)(idx % C);
    long r = idx / C;
    int j = (int)(r % N); long r2 = r / N;
    int i = (int)(r2 % N); int b = (int)(r2 / N);
    float v = fabsf(X[((long)b * N + i) * C + c] - X[((long)b * N + j) * C + c]);
    H[idx] = (half_t)v;
  }
}

// per-column sum / sum-of-squares over M rows (one block per column)
__global__ void col_stats_kernel(const half_t* __restrict__ H,
                                 float* __restrict__ sum, float* __restrict__ sumsq,
                                 long M, int C)
{
  int c = blockIdx.x;
  float s = 0.f, q = 0.f;
  for (long m = threadIdx.x; m < M; m += blockDim.x) {
    float v = (float)H[m * C + c];
    s += v; q += v * v;
  }
  __shared__ float ss[256], qq[256];
  ss[threadIdx.x] = s; qq[threadIdx.x] = q;
  __syncthreads();
  for (int o = 128; o > 0; o >>= 1) {
    if ((int)threadIdx.x < o) { ss[threadIdx.x] += ss[threadIdx.x + o]; qq[threadIdx.x] += qq[threadIdx.x + o]; }
    __syncthreads();
  }
  if (threadIdx.x == 0) { sum[c] = ss[0]; sumsq[c] = qq[0]; }
}

// batch-norm (runtime stats) + leaky relu, in place on f16 activations
__global__ void norm_leaky_kernel(half_t* __restrict__ H,
                                  const float* __restrict__ sum, const float* __restrict__ sumsq,
                                  const float* __restrict__ g, const float* __restrict__ beta,
                                  long M, int C)
{
  long total = M * C;
  float invM = 1.0f / (float)M;
  for (long idx = gtid(); idx < total; idx += gstride()) {
    int c = (int)(idx % C);
    float mu  = sum[c] * invM;
    float var = sumsq[c] * invM - mu * mu;
    float x = (float)H[idx];
    float y = (x - mu) * rsqrtf(var + 1e-5f) * g[c] + beta[c];
    H[idx] = (half_t)(y > 0.f ? y : 0.01f * y);
  }
}

// logits[m] = dot(H[m,:C], wl) + bl
__global__ void final_lin_kernel(const half_t* __restrict__ H, const half_t* __restrict__ wl,
                                 const float* __restrict__ bl, float* __restrict__ out,
                                 long M, int C)
{
  for (long m = gtid(); m < M; m += gstride()) {
    const half_t* h = H + m * C;
    float s = 0.f;
    for (int c = 0; c < C; ++c) s += (float)h[c] * (float)wl[c];
    out[m] = s + bl[0];
  }
}

// softmax over rows of length N; write f16 adjacency zero-padded to Kpad
__global__ void softmax_row_kernel(const float* __restrict__ logits, half_t* __restrict__ A16,
                                   int N, int Kpad)
{
  long r = blockIdx.x;             // b*N + i
  const float* row = logits + r * N;
  __shared__ float red[256];
  float mx = -3.4e38f;
  for (int j = threadIdx.x; j < N; j += blockDim.x) mx = fmaxf(mx, row[j]);
  red[threadIdx.x] = mx; __syncthreads();
  for (int o = 128; o > 0; o >>= 1) { if ((int)threadIdx.x < o) red[threadIdx.x] = fmaxf(red[threadIdx.x], red[threadIdx.x + o]); __syncthreads(); }
  mx = red[0]; __syncthreads();
  float s = 0.f;
  for (int j = threadIdx.x; j < N; j += blockDim.x) s += expf(row[j] - mx);
  red[threadIdx.x] = s; __syncthreads();
  for (int o = 128; o > 0; o >>= 1) { if ((int)threadIdx.x < o) red[threadIdx.x] += red[threadIdx.x + o]; __syncthreads(); }
  float inv = 1.0f / red[0];
  half_t* arow = A16 + r * Kpad;
  for (int j = threadIdx.x; j < Kpad; j += blockDim.x)
    arow[j] = (half_t)(j < N ? expf(row[j] - mx) * inv : 0.0f);
}

// Xt[b,c,j] = X[b,j,c]  (f32 -> f16, zero-padded to Kpad)
__global__ void transpose_x_kernel(const float* __restrict__ X, half_t* __restrict__ Xt,
                                   int N, int C, int Kpad)
{
  long total = (long)NB * C * Kpad;
  for (long idx = gtid(); idx < total; idx += gstride()) {
    int j = (int)(idx % Kpad);
    int c = (int)((idx / Kpad) % C);
    int b = (int)(idx / ((long)Kpad * C));
    Xt[idx] = (half_t)(j < N ? X[((long)b * N + j) * C + c] : 0.0f);
  }
}

__global__ void f32_to_f16_kernel(const float* __restrict__ src, half_t* __restrict__ dst, long n)
{
  for (long i = gtid(); i < n; i += gstride()) dst[i] = (half_t)src[i];
}

// last NQ rows: Xnew = 0.2*Xnew + 0.8*Xold (in place on Xnew)
__global__ void blend_kernel(float* __restrict__ Xn, const float* __restrict__ Xo,
                             int N, int C)
{
  long total = (long)NB * NQV * C;
  for (long idx = gtid(); idx < total; idx += gstride()) {
    int c = (int)(idx % C);
    int q = (int)((idx / C) % NQV);
    int b = (int)(idx / ((long)C * NQV));
    long o = ((long)b * N + (N - NQV + q)) * C + c;
    Xn[o] = 0.2f * Xn[o] + 0.8f * Xo[o];
  }
}

// scores[b,n] = sigmoid((X[b,n]·w + b0)/100)
__global__ void scores_kernel(const float* __restrict__ X, const float* __restrict__ w,
                              const float* __restrict__ b0, float* __restrict__ s,
                              int N, int C)
{
  long total = (long)NB * N;
  for (long i = gtid(); i < total; i += gstride()) {
    const float* x = X + i * C;
    float d = 0.f;
    for (int c = 0; c < C; ++c) d += x[c] * w[c];
    float v = (d + b0[0]) * 0.01f;
    s[i] = 1.0f / (1.0f + expf(-v));
  }
}

// iterative arg-max top-k per batch (first-index on ties == jax top_k stability)
__global__ void topk_kernel(const float* __restrict__ scores, int* __restrict__ sidx,
                            float* __restrict__ svals, int N, int ns, int keep)
{
  if (threadIdx.x != 0) return;
  int b = blockIdx.x;
  const float* sc = scores + (long)b * N;
  bool used[256];
  for (int i = 0; i < ns; ++i) used[i] = false;
  for (int t = 0; t < keep; ++t) {
    int best = 0; float bv = -3.4e38f;
    for (int i = 0; i < ns; ++i)
      if (!used[i] && sc[i] > bv) { bv = sc[i]; best = i; }
    used[best] = true;
    sidx[(long)b * keep + t] = best;
    svals[(long)b * keep + t] = bv;
  }
}

// newX[b,t,c] = X[b, idx(t), c] * val(t)
__global__ void gather_kernel(const float* __restrict__ X, const float* __restrict__ scores,
                              const int* __restrict__ sidx, const float* __restrict__ svals,
                              float* __restrict__ Xn, int Nold, int Nnew, int C,
                              int keep, int ns)
{
  long total = (long)NB * Nnew * C;
  for (long idx = gtid(); idx < total; idx += gstride()) {
    int c = (int)(idx % C);
    long r = idx / C;
    int t = (int)(r % Nnew);
    int b = (int)(r / Nnew);
    int src; float val;
    if (t < keep) { src = sidx[(long)b * keep + t]; val = svals[(long)b * keep + t]; }
    else          { src = ns + (t - keep);          val = scores[(long)b * Nold + src]; }
    Xn[idx] = X[((long)b * Nold + src) * C + c] * val;
  }
}

__global__ void logsoftmax_kernel(const float* __restrict__ in, float* __restrict__ out,
                                  long rows, int C)
{
  for (long r = gtid(); r < rows; r += gstride()) {
    const float* x = in + r * C;
    float mx = x[0];
    for (int c = 1; c < C; ++c) mx = fmaxf(mx, x[c]);
    float s = 0.f;
    for (int c = 0; c < C; ++c) s += expf(x[c] - mx);
    float l = logf(s);
    for (int c = 0; c < C; ++c) out[r * C + c] = x[c] - mx - l;
  }
}

// ---------------------------------------------------------------------------
// Host orchestration
// ---------------------------------------------------------------------------
struct MlpDev {
  const half_t *w0, *w1, *w2, *w3, *wl;
  const float *g0, *g1, *g2, *g3;
  const float *b0, *b1, *b2, *b3, *bl;
};

// Flattened input indices assuming jax-style sorted-key tree flattening:
// A_init, X, then params keys alphabetically; lin = [b,w],
// mlp = [b0,b1,b2,b3,bl,g0,g1,g2,g3,w0,w1,w2,w3,wl]
enum {
  IN_X = 1,
  BOT_GCN_B = 2, BOT_GCN_W = 3,
  BOT_MLP = 4,
  DG1_0_B = 18, DG1_0_W = 19, DG1_1_B = 20, DG1_1_W = 21, DG1_2_B = 22, DG1_2_W = 23,
  DG2_0_B = 24, DG2_0_W = 25, DG2_1_B = 26, DG2_1_W = 27, DG2_2_B = 28, DG2_2_W = 29,
  DMLP_0 = 30, DMLP_1 = 44, DMLP_2 = 58,
  OUT_GCN_B = 72, OUT_GCN_W = 73,
  OUT_MLP = 74,
  POOL_0_B = 88, POOL_0_W = 89, POOL_1_B = 90, POOL_1_W = 91, POOL_2_B = 92, POOL_2_W = 93,
  START_GCN_B = 94, START_GCN_W = 95,
  START_MLP = 96
};

extern "C" void kernel_launch(void* const* d_in, const int* in_sizes, int n_in,
                              void* d_out, int out_size, void* d_ws, size_t ws_size,
                              hipStream_t stream)
{
  (void)in_sizes; (void)n_in; (void)out_size; (void)ws_size;

  const long MMAX = 8L * 250 * 250;   // 500000 pairwise rows

  // ---- workspace carve-up ------------------------------------------------
  size_t off = 0;
  auto alloc = [&](size_t bytes) -> void* {
    void* p = (char*)d_ws + off;
    off = (off + bytes + 255) & ~(size_t)255;
    return p;
  };
  half_t* hA     = (half_t*)alloc(MMAX * 192 * sizeof(half_t));
  half_t* hB     = (half_t*)alloc(MMAX * 192 * sizeof(half_t));
  float*  logits = (float*) alloc(MMAX * sizeof(float));
  half_t* a16    = (half_t*)alloc(8L * 250 * KPAD * sizeof(half_t));
  half_t* xt16   = (half_t*)alloc(8L * CH * KPAD * sizeof(half_t));
  half_t* tmp16  = (half_t*)alloc(8L * 250 * CH * sizeof(half_t));
  half_t* xh16   = (half_t*)alloc(8L * 250 * CH * sizeof(half_t));
  float*  Xbuf0  = (float*) alloc(8L * 250 * CH * sizeof(float));
  float*  Xbuf1  = (float*) alloc(8L * 250 * CH * sizeof(float));
  float*  Xbuf2  = (float*) alloc(8L * 250 * CH * sizeof(float));
  float*  xo     = (float*) alloc(8L * 250 * 5 * sizeof(float));
  float*  ssum   = (float*) alloc(192 * sizeof(float));
  float*  ssq    = (float*) alloc(192 * sizeof(float));
  float*  sc     = (float*) alloc(8L * 250 * sizeof(float));
  int*    sidx   = (int*)   alloc(8L * 160 * sizeof(int));
  float*  svals  = (float*) alloc(8L * 160 * sizeof(float));
  half_t* whbase = (half_t*)alloc(4 * 1024 * 1024);
  float*  Xbuf[3] = { Xbuf0, Xbuf1, Xbuf2 };

  auto nblk = [](long n) -> unsigned { long b = (n + 255) / 256; return b > 0 ? (unsigned)b : 1u; };

  // ---- f32 -> f16 weight staging ----------------------------------------
  size_t woff = 0;
  auto cvtw = [&](int idx, long n) -> const half_t* {
    half_t* dst = whbase + woff;
    woff = (woff + n + 127) & ~(size_t)127;
    f32_to_f16_kernel<<<nblk(n), 256, 0, stream>>>((const float*)d_in[idx], dst, n);
    return dst;
  };
  auto mkMlp = [&](int base) -> MlpDev {
    MlpDev m;
    m.b0 = (const float*)d_in[base + 0]; m.b1 = (const float*)d_in[base + 1];
    m.b2 = (const float*)d_in[base + 2]; m.b3 = (const float*)d_in[base + 3];
    m.bl = (const float*)d_in[base + 4];
    m.g0 = (const float*)d_in[base + 5]; m.g1 = (const float*)d_in[base + 6];
    m.g2 = (const float*)d_in[base + 7]; m.g3 = (const float*)d_in[base + 8];
    m.w0 = cvtw(base + 9, 192 * 128);
    m.w1 = cvtw(base + 10, 192 * 192);
    m.w2 = cvtw(base + 11, 96 * 192);
    m.w3 = cvtw(base + 12, 96 * 96);
    m.wl = cvtw(base + 13, 96);
    return m;
  };

  MlpDev startMlp  = mkMlp(START_MLP);
  MlpDev downMlp[3] = { mkMlp(DMLP_0), mkMlp(DMLP_1), mkMlp(DMLP_2) };
  MlpDev botMlp    = mkMlp(BOT_MLP);
  MlpDev outMlp    = mkMlp(OUT_MLP);

  const half_t* startW  = cvtw(START_GCN_W, 128 * 128);
  const half_t* botW    = cvtw(BOT_GCN_W,   128 * 128);
  const half_t* outW    = cvtw(OUT_GCN_W,   5 * 128);
  const half_t* d1W[3]  = { cvtw(DG1_0_W, 128 * 128), cvtw(DG1_1_W, 128 * 128), cvtw(DG1_2_W, 128 * 128) };
  const half_t* d2W[3]  = { cvtw(DG2_0_W, 5 * 128),   cvtw(DG2_1_W, 5 * 128),   cvtw(DG2_2_W, 5 * 128) };
  const float*  startB  = (const float*)d_in[START_GCN_B];
  const float*  botB    = (const float*)d_in[BOT_GCN_B];
  const float*  outB    = (const float*)d_in[OUT_GCN_B];
  const float*  d1B[3]  = { (const float*)d_in[DG1_0_B], (const float*)d_in[DG1_1_B], (const float*)d_in[DG1_2_B] };
  const float*  d2B[3]  = { (const float*)d_in[DG2_0_B], (const float*)d_in[DG2_1_B], (const float*)d_in[DG2_2_B] };
  const float*  poolW[3] = { (const float*)d_in[POOL_0_W], (const float*)d_in[POOL_1_W], (const float*)d_in[POOL_2_W] };
  const float*  poolB[3] = { (const float*)d_in[POOL_0_B], (const float*)d_in[POOL_1_B], (const float*)d_in[POOL_2_B] };

  // ---- GEMM launcher: pick N-register-blocking by layer width ------------
  auto gemm = [&](const half_t* A, const half_t* W, const float* bias,
                  float* Cf, half_t* Chh,
                  long M, int Nn, int K, int lda, int ldw, int ldc,
                  int batch, long sA, long sW, long sC) {
    int tilesM = (int)((M + 15) / 16);
    int tilesN = (Nn + 15) / 16;
    int NT = (tilesN % 4 == 0) ? 4 : (tilesN % 2 == 0) ? 2 : 1;
    long strips = (long)tilesM * ((tilesN + NT - 1) / NT);
    dim3 g((unsigned)((strips + 7) / 8), (unsigned)batch);
    if (NT == 4)
      wmma_gemm_kernel<4><<<g, 256, 0, stream>>>(A, W, bias, Cf, Chh, (int)M, Nn, K, lda, ldw, ldc, sA, sW, sC);
    else if (NT == 2)
      wmma_gemm_kernel<2><<<g, 256, 0, stream>>>(A, W, bias, Cf, Chh, (int)M, Nn, K, lda, ldw, ldc, sA, sW, sC);
    else
      wmma_gemm_kernel<1><<<g, 256, 0, stream>>>(A, W, bias, Cf, Chh, (int)M, Nn, K, lda, ldw, ldc, sA, sW, sC);
  };

  // ---- _mlp_adj: X (B,N,128) f32 -> adjacency a16 (B,N,KPAD) f16 ---------
  auto run_mlp = [&](const float* X, int N, const MlpDev& m) {
    long M = 8L * N * N;
    pair_abs_kernel<<<nblk(M * 128), 256, 0, stream>>>(X, hA, N, 128);

    gemm(hA, m.w0, nullptr, nullptr, hB, M, 192, 128, 128, 128, 192, 1, 0, 0, 0);
    col_stats_kernel<<<192, 256, 0, stream>>>(hB, ssum, ssq, M, 192);
    norm_leaky_kernel<<<nblk(M * 192), 256, 0, stream>>>(hB, ssum, ssq, m.g0, m.b0, M, 192);

    gemm(hB, m.w1, nullptr, nullptr, hA, M, 192, 192, 192, 192, 192, 1, 0, 0, 0);
    col_stats_kernel<<<192, 256, 0, stream>>>(hA, ssum, ssq, M, 192);
    norm_leaky_kernel<<<nblk(M * 192), 256, 0, stream>>>(hA, ssum, ssq, m.g1, m.b1, M, 192);

    gemm(hA, m.w2, nullptr, nullptr, hB, M, 96, 192, 192, 192, 96, 1, 0, 0, 0);
    col_stats_kernel<<<96, 256, 0, stream>>>(hB, ssum, ssq, M, 96);
    norm_leaky_kernel<<<nblk(M * 96), 256, 0, stream>>>(hB, ssum, ssq, m.g2, m.b2, M, 96);

    gemm(hB, m.w3, nullptr, nullptr, hA, M, 96, 96, 96, 96, 96, 1, 0, 0, 0);
    col_stats_kernel<<<96, 256, 0, stream>>>(hA, ssum, ssq, M, 96);
    norm_leaky_kernel<<<nblk(M * 96), 256, 0, stream>>>(hA, ssum, ssq, m.g3, m.b3, M, 96);

    final_lin_kernel<<<nblk(M), 256, 0, stream>>>(hA, m.wl, m.bl, logits, M, 96);
    softmax_row_kernel<<<8 * N, 256, 0, stream>>>(logits, a16, N, KPAD);
  };

  // ---- _gcn: Xnew = (A@X) @ W^T + b  (optional blend of last NQ rows) ----
  auto run_gcn = [&](const float* X, int N, const half_t* Wh, const float* bias, int Fo,
                     bool blend, const float* Xold, float* Xnew) {
    transpose_x_kernel<<<nblk(8L * CH * KPAD), 256, 0, stream>>>(X, xt16, N, CH, KPAD);
    // tmp(b) = A(b) @ X(b) : batched WMMA GEMM, K = KPAD (zero-padded)
    gemm(a16, xt16, nullptr, nullptr, tmp16,
         N, CH, KPAD, KPAD, KPAD, CH, 8, (long)N * KPAD, (long)CH * KPAD, (long)N * CH);
    // Xnew = tmp @ W^T + b : flatten batches (tmp rows are packed)
    gemm(tmp16, Wh, bias, Xnew, nullptr, 8L * N, Fo, 128, 128, 128, Fo, 1, 0, 0, 0);
    if (blend) blend_kernel<<<nblk(8L * NQV * Fo), 256, 0, stream>>>(Xnew, Xold, N, Fo);
  };

  // ======================= pipeline ======================================
  const float* Xin = (const float*)d_in[IN_X];
  float* outp = (float*)d_out;
  long outOff = 0;
  const double ks[3] = { 0.6, 0.5, 0.5 };

  // start block
  run_mlp(Xin, 250, startMlp);
  run_gcn(Xin, 250, startW, startB, 128, false, nullptr, Xbuf[0]);
  const float* cur = Xbuf[0];
  int N = 250;
  int freeA = 1, freeB = 2;

  // down blocks
  for (int i = 0; i < 3; ++i) {
    const float* Xold = cur;
    run_mlp(cur, N, downMlp[i]);
    float* Xg = Xbuf[freeA];
    run_gcn(cur, N, d1W[i], d1B[i], 128, true, Xold, Xg);

    // Xo = Xg @ W2^T + b2 -> log_softmax -> output i
    f32_to_f16_kernel<<<nblk(8L * N * CH), 256, 0, stream>>>(Xg, xh16, 8L * N * CH);
    gemm(xh16, d2W[i], d2B[i], xo, nullptr, 8L * N, 5, 128, 128, 128, 5, 1, 0, 0, 0);
    logsoftmax_kernel<<<nblk(8L * N), 256, 0, stream>>>(xo, outp + outOff, 8L * N, 5);
    outOff += 8L * N * 5;

    // pool (newA is dead in the reference; only newX matters)
    scores_kernel<<<nblk(8L * N), 256, 0, stream>>>(Xg, poolW[i], poolB[i], sc, N, CH);
    int ns = N - NQV;
    int keep = (int)(ks[i] * (double)ns);
    topk_kernel<<<8, 32, 0, stream>>>(sc, sidx, svals, N, ns, keep);
    int N2 = keep + NQV;
    float* Xp = Xbuf[freeB];
    gather_kernel<<<nblk(8L * N2 * CH), 256, 0, stream>>>(Xg, sc, sidx, svals, Xp, N, N2, CH, keep, ns);

    int curIdx = (cur == Xbuf[0]) ? 0 : (cur == Xbuf[1]) ? 1 : 2;
    cur = Xp; N = N2;
    int nA = curIdx, nB = freeA;
    freeA = nA; freeB = nB;
  }

  // bottom block (blend, Fo=128)
  {
    const float* Xold = cur;
    run_mlp(cur, N, botMlp);
    float* Xg = Xbuf[freeA];
    run_gcn(cur, N, botW, botB, 128, true, Xold, Xg);
    cur = Xg;
  }

  // out block (Fo = NC = 5, no blend) -> log_softmax -> output 3
  run_mlp(cur, N, outMlp);
  run_gcn(cur, N, outW, outB, 5, false, nullptr, xo);
  logsoftmax_kernel<<<nblk(8L * N), 256, 0, stream>>>(xo, outp + outOff, 8L * N, 5);
}
